// MoELayer_76012331204735
// MI455X (gfx1250) — matrix operand order repaired and checked by reading, not compiled, not run on hardware
//
#include <hip/hip_runtime.h>
#include <hip/hip_bf16.h>
#include <math.h>

// Problem constants (match reference)
#define B_  4
#define L_  2048
#define D_  1024
#define E_  8
#define H_  4096
#define K_  2
#define BL  (B_*L_)   // 8192 tokens

typedef __attribute__((ext_vector_type(16))) __bf16 v16bf;
typedef __attribute__((ext_vector_type(8)))  float  v8f;

union Frag16 {
    v16bf v;
    uint4 q[2];
    unsigned short u[16];
};

__device__ __forceinline__ unsigned short f2bf(float f) {
    unsigned int u = __float_as_uint(f);
    unsigned int r = u + 0x7FFFu + ((u >> 16) & 1u);   // round-to-nearest-even
    return (unsigned short)(r >> 16);
}

__device__ __forceinline__ float gelu_tanh(float x) {
    float x3 = x * x * x;
    return 0.5f * x * (1.0f + tanhf(0.7978845608028654f * (x + 0.044715f * x3)));
}

// ---------------------------------------------------------------------------
// small init: zero per-expert counters + prob sums
// ---------------------------------------------------------------------------
__global__ void moe_init_small(int* counts, float* probsum) {
    int t = threadIdx.x;
    if (t < E_) { counts[t] = 0; probsum[t] = 0.0f; }
}

// ---------------------------------------------------------------------------
// zero the main output (accumulated into by expert kernel via atomics)
// ---------------------------------------------------------------------------
__global__ void moe_zero_out(float* __restrict__ out) {
    size_t i = ((size_t)blockIdx.x * 256 + threadIdx.x) * 4;
    float4 z = {0.f, 0.f, 0.f, 0.f};
    *(float4*)(out + i) = z;
}

// ---------------------------------------------------------------------------
// x: f32 -> bf16 elementwise
// ---------------------------------------------------------------------------
__global__ void moe_convert_x(const float* __restrict__ x, unsigned short* __restrict__ xb) {
    size_t i = ((size_t)blockIdx.x * 256 + threadIdx.x) * 4;
    unsigned short r0 = f2bf(x[i + 0]);
    unsigned short r1 = f2bf(x[i + 1]);
    unsigned short r2 = f2bf(x[i + 2]);
    unsigned short r3 = f2bf(x[i + 3]);
    uint2 packed;
    packed.x = (unsigned)r0 | ((unsigned)r1 << 16);
    packed.y = (unsigned)r2 | ((unsigned)r3 << 16);
    *(uint2*)(xb + i) = packed;
}

// ---------------------------------------------------------------------------
// batched transpose + convert: in (e, R, C) f32  ->  out (e, C, R) bf16
// Makes WMMA B-fragments contiguous-in-K (two b128 loads per lane).
// ---------------------------------------------------------------------------
__global__ void moe_transpose_bf16(const float* __restrict__ in,
                                   unsigned short* __restrict__ out,
                                   int R, int C) {
    __shared__ float tile[32][33];
    const float* inb = in + (size_t)blockIdx.z * R * C;
    unsigned short* outb = out + (size_t)blockIdx.z * R * C;
    int r0 = blockIdx.y * 32, c0 = blockIdx.x * 32;
    int tx = threadIdx.x, ty = threadIdx.y;            // 32 x 8
    #pragma unroll
    for (int i = 0; i < 32; i += 8)
        tile[ty + i][tx] = inb[(size_t)(r0 + ty + i) * C + (c0 + tx)];
    __syncthreads();
    #pragma unroll
    for (int i = 0; i < 32; i += 8)
        outb[(size_t)(c0 + ty + i) * R + (r0 + tx)] = f2bf(tile[tx][ty + i]);
}

// ---------------------------------------------------------------------------
// Router: one wave32 per token. logits = x.gate_w, softmax(E=8), top-2,
// renormalize, emit per-token coef row, compacted per-expert token lists,
// and prob sums / counts for the aux loss.
// ---------------------------------------------------------------------------
__global__ __launch_bounds__(256, 1)
void moe_router(const float* __restrict__ x, const float* __restrict__ gate_w,
                float* __restrict__ coef, int* __restrict__ lists,
                int* __restrict__ counts, float* __restrict__ probsum) {
    int wv   = threadIdx.x >> 5;
    int lane = threadIdx.x & 31;
    int tok  = blockIdx.x * 8 + wv;
    if (tok >= BL) return;

    const float* xr = x + (size_t)tok * D_;
    float acc[E_];
    #pragma unroll
    for (int e = 0; e < E_; e++) acc[e] = 0.f;

    for (int d = lane; d < D_; d += 32) {
        float xv = xr[d];
        const float4* gp = (const float4*)(gate_w + (size_t)d * E_);
        float4 g0 = gp[0], g1 = gp[1];
        acc[0] += xv * g0.x; acc[1] += xv * g0.y; acc[2] += xv * g0.z; acc[3] += xv * g0.w;
        acc[4] += xv * g1.x; acc[5] += xv * g1.y; acc[6] += xv * g1.z; acc[7] += xv * g1.w;
    }
    #pragma unroll
    for (int off = 16; off > 0; off >>= 1) {
        #pragma unroll
        for (int e = 0; e < E_; e++) acc[e] += __shfl_down(acc[e], off, 32);
    }
    if (lane == 0) {
        float m = acc[0];
        #pragma unroll
        for (int e = 1; e < E_; e++) m = fmaxf(m, acc[e]);
        float p[E_], s = 0.f;
        #pragma unroll
        for (int e = 0; e < E_; e++) { p[e] = __expf(acc[e] - m); s += p[e]; }
        float inv = 1.0f / s;
        int i1 = 0;
        #pragma unroll
        for (int e = 1; e < E_; e++) if (p[e] > p[i1]) i1 = e;
        int i2 = (i1 == 0) ? 1 : 0;
        #pragma unroll
        for (int e = 0; e < E_; e++) if (e != i1 && p[e] > p[i2]) i2 = e;
        float p1 = p[i1] * inv, p2 = p[i2] * inv;
        float wsum = p1 + p2;
        #pragma unroll
        for (int e = 0; e < E_; e++) {
            float pe = p[e] * inv;
            atomicAdd(&probsum[e], pe);
            coef[(size_t)tok * E_ + e] = (e == i1) ? (p1 / wsum)
                                       : (e == i2) ? (p2 / wsum) : 0.0f;
        }
        int pos1 = atomicAdd(&counts[i1], 1); lists[i1 * BL + pos1] = tok;
        int pos2 = atomicAdd(&counts[i2], 1); lists[i2 * BL + pos2] = tok;
    }
}

// ---------------------------------------------------------------------------
// Fused expert FFN: per block = (expert e, tile of 32 routed tokens).
// 8 waves. For each 64-wide H chunk:
//   phase A: wave w computes one 16x16 tile of h = gelu(X @ W1 + b1)  (K=1024)
//   phase B: wave w accumulates its 128-col slice of Y += h @ W2      (K=64)
// X is staged into LDS with GLOBAL_LOAD_ASYNC_TO_LDS_B128 (ASYNCcnt path).
// Epilogue: Y = (Y + b2) * coef[token], scattered with f32 atomics.
// ---------------------------------------------------------------------------
__global__ __launch_bounds__(256, 1)
void moe_expert_ffn(const unsigned short* __restrict__ xb,   // (BL, D)  bf16
                    const unsigned short* __restrict__ w1t,  // (E, H, D) bf16 (transposed)
                    const unsigned short* __restrict__ w2t,  // (E, D, H) bf16 (transposed)
                    const float* __restrict__ b1,            // (E, H)
                    const float* __restrict__ b2,            // (E, D)
                    const float* __restrict__ coef,          // (BL, E)
                    const int*   __restrict__ lists,         // (E, BL)
                    const int*   __restrict__ counts,        // (E)
                    float* __restrict__ out)                 // (BL, D)
{
    __shared__ unsigned short Xs[32][1032];   // 32 token rows x D, +8 pad (bank-conflict free)
    __shared__ unsigned short Hs[32][72];     // 32 x 64 h-chunk, +8 pad
    __shared__ int   toks[32];
    __shared__ float cfs[32];

    const int e   = blockIdx.y;
    const int cnt = counts[e];
    const int t0  = blockIdx.x * 32;
    if (t0 >= cnt) return;

    const int tid  = threadIdx.x;
    const int w    = tid >> 5;       // wave 0..7
    const int lane = tid & 31;
    const int g    = lane >> 4;      // half-wave group (K-split in A/B layouts)
    const int ln   = lane & 15;      // M (A) / N (B,C) index

    if (tid < 32) {
        int i = t0 + tid;
        if (i < cnt) {
            int t = lists[e * BL + i];
            toks[tid] = t;
            cfs[tid]  = coef[(size_t)t * E_ + e];
        } else { toks[tid] = 0; cfs[tid] = 0.0f; }
    }
    __syncthreads();

    // Stage gathered X rows into LDS with async copies (no VGPR round-trip).
    // GVS form: dsaddr VGPR, 32-bit voffset VGPR, 64-bit SGPR base.
    {
        int row = tid >> 3;          // 0..31
        int seg = tid & 7;           // 128 bf16 per thread
        unsigned gbyte0 = ((unsigned)toks[row] * D_ + (unsigned)seg * 128u) * 2u;
        unsigned lds0   = (unsigned)(size_t)(&Xs[row][seg * 128]);  // addr[31:0] = LDS offset
        #pragma unroll
        for (int i = 0; i < 8; i++) {
            unsigned ldsa = lds0   + (unsigned)(i * 16);
            unsigned gofs = gbyte0 + (unsigned)(i * 16);
            asm volatile("global_load_async_to_lds_b128 %0, %1, %2"
                         :: "v"(ldsa), "v"(gofs), "s"(xb)
                         : "memory");
        }
        asm volatile("s_wait_asynccnt 0x0" ::: "memory");
    }
    __syncthreads();

    const int mi = w & 1;            // up-proj tile row  (0..1)
    const int ni = w >> 1;           // up-proj tile col  (0..3)
    const int c0 = w * 128;          // this wave's output-column window

    v8f Y[2][8];
    #pragma unroll
    for (int a = 0; a < 2; a++)
        #pragma unroll
        for (int b = 0; b < 8; b++)
            #pragma unroll
            for (int r = 0; r < 8; r++) Y[a][b][r] = 0.0f;

    const int rowA = mi * 16 + ln;

    for (int hc = 0; hc < H_; hc += 64) {
        // -------- phase A: h(32x64) tile per wave --------
        const unsigned short* w1row =
            w1t + ((size_t)e * H_ + hc + ni * 16 + ln) * D_ + 16 * g;
        // pull next chunk's weights toward the WGP while this chunk computes
        if (hc + 64 < H_) {
            __builtin_prefetch((const void*)(w1row + (size_t)64 * D_), 0, 0);
            __builtin_prefetch((const void*)(w2t + ((size_t)e * D_ + c0 + ln) * H_ + hc + 64), 0, 0);
        }

        v8f hacc = {0.f, 0.f, 0.f, 0.f, 0.f, 0.f, 0.f, 0.f};
        Frag16 bcur;
        {
            const uint4* bp = (const uint4*)(w1row);
            bcur.q[0] = bp[0]; bcur.q[1] = bp[1];
        }
        #pragma unroll 4
        for (int k0 = 0; k0 < D_; k0 += 32) {
            Frag16 a, bnext;
            if (k0 + 32 < D_) {                     // double-buffer B fragment
                const uint4* bp = (const uint4*)(w1row + k0 + 32);
                bnext.q[0] = bp[0]; bnext.q[1] = bp[1];
            }
            a.q[0] = *(const uint4*)(&Xs[rowA][k0 + 16 * g]);
            a.q[1] = *(const uint4*)(&Xs[rowA][k0 + 16 * g + 8]);
            hacc = __builtin_amdgcn_wmma_f32_16x16x32_bf16(
                false, a.v, false, bcur.v, (short)0, hacc, false, false);
            if (k0 + 32 < D_) bcur = bnext;
        }
        float b1v = b1[(size_t)e * H_ + hc + ni * 16 + ln];
        #pragma unroll
        for (int r = 0; r < 8; r++) {
            float hv = gelu_tanh(hacc[r] + b1v);
            Hs[mi * 16 + 8 * g + r][ni * 16 + ln] = f2bf(hv);
        }
        __syncthreads();

        // -------- phase B: Y[:, c0:c0+128] += h @ W2[hc:hc+64, :] --------
        #pragma unroll
        for (int ki = 0; ki < 2; ki++) {
            Frag16 a0, a1;
            a0.q[0] = *(const uint4*)(&Hs[ln][ki * 32 + 16 * g]);
            a0.q[1] = *(const uint4*)(&Hs[ln][ki * 32 + 16 * g + 8]);
            a1.q[0] = *(const uint4*)(&Hs[16 + ln][ki * 32 + 16 * g]);
            a1.q[1] = *(const uint4*)(&Hs[16 + ln][ki * 32 + 16 * g + 8]);

            const unsigned short* w2base =
                w2t + ((size_t)e * D_ + c0 + ln) * H_ + hc + ki * 32 + 16 * g;
            Frag16 b0;
            {
                const uint4* bp = (const uint4*)(w2base);
                b0.q[0] = bp[0]; b0.q[1] = bp[1];
            }
            #pragma unroll
            for (int n2 = 0; n2 < 8; n2++) {
                Frag16 bn;
                if (n2 < 7) {                       // double-buffer B fragment
                    const uint4* bp = (const uint4*)(w2base + (size_t)(n2 + 1) * 16 * H_);
                    bn.q[0] = bp[0]; bn.q[1] = bp[1];
                }
                Y[0][n2] = __builtin_amdgcn_wmma_f32_16x16x32_bf16(
                    false, a0.v, false, b0.v, (short)0, Y[0][n2], false, false);
                Y[1][n2] = __builtin_amdgcn_wmma_f32_16x16x32_bf16(
                    false, a1.v, false, b0.v, (short)0, Y[1][n2], false, false);
                if (n2 < 7) b0 = bn;
            }
        }
        __syncthreads();
    }

    // -------- epilogue: bias, routing coefficient, atomic scatter --------
    #pragma unroll
    for (int m2 = 0; m2 < 2; m2++) {
        #pragma unroll
        for (int n2 = 0; n2 < 8; n2++) {
            float b2v = b2[(size_t)e * D_ + c0 + n2 * 16 + ln];
            #pragma unroll
            for (int r = 0; r < 8; r++) {
                int row = m2 * 16 + 8 * g + r;
                float cf = cfs[row];
                if (cf != 0.0f) {
                    float val = (Y[m2][n2][r] + b2v) * cf;
                    atomicAdd(&out[(size_t)toks[row] * D_ + c0 + n2 * 16 + ln], val);
                }
            }
        }
    }
}

// ---------------------------------------------------------------------------
// aux loss: E * sum_e f_e * P_e
// ---------------------------------------------------------------------------
__global__ void moe_aux(const int* __restrict__ counts,
                        const float* __restrict__ probsum,
                        float* __restrict__ out_aux) {
    if (threadIdx.x == 0 && blockIdx.x == 0) {
        float s = 0.f;
        for (int e = 0; e < E_; e++) {
            float f = (float)counts[e] / (float)(BL * K_);
            float P = probsum[e] / (float)BL;
            s += f * P;
        }
        *out_aux = (float)E_ * s;
    }
}

// ---------------------------------------------------------------------------
extern "C" void kernel_launch(void* const* d_in, const int* in_sizes, int n_in,
                              void* d_out, int out_size, void* d_ws, size_t ws_size,
                              hipStream_t stream) {
    (void)in_sizes; (void)n_in; (void)out_size; (void)ws_size;
    const float* x      = (const float*)d_in[0];
    const float* gate_w = (const float*)d_in[1];
    const float* w1     = (const float*)d_in[2];
    const float* b1     = (const float*)d_in[3];
    const float* w2     = (const float*)d_in[4];
    const float* b2     = (const float*)d_in[5];
    float* out = (float*)d_out;

    char* ws = (char*)d_ws;
    size_t o = 0;
    auto take = [&](size_t bytes) -> char* {
        char* p = ws + o;
        o += (bytes + 255) & ~(size_t)255;
        return p;
    };
    unsigned short* xb      = (unsigned short*)take((size_t)BL * D_ * 2);      // 16 MB
    unsigned short* w1t     = (unsigned short*)take((size_t)E_ * H_ * D_ * 2); // 64 MB
    unsigned short* w2t     = (unsigned short*)take((size_t)E_ * D_ * H_ * 2); // 64 MB
    float*          coef    = (float*)take((size_t)BL * E_ * 4);
    int*            lists   = (int*)take((size_t)E_ * BL * 4);
    int*            counts  = (int*)take(256);
    float*          probsum = (float*)take(256);

    moe_init_small<<<1, 64, 0, stream>>>(counts, probsum);
    moe_zero_out<<<(BL * D_) / (256 * 4), 256, 0, stream>>>(out);
    moe_convert_x<<<(BL * D_) / (256 * 4), 256, 0, stream>>>(x, xb);
    // w1 (E,D,H) -> w1t (E,H,D)
    moe_transpose_bf16<<<dim3(H_ / 32, D_ / 32, E_), dim3(32, 8), 0, stream>>>(w1, w1t, D_, H_);
    // w2 (E,H,D) -> w2t (E,D,H)
    moe_transpose_bf16<<<dim3(D_ / 32, H_ / 32, E_), dim3(32, 8), 0, stream>>>(w2, w2t, H_, D_);
    moe_router<<<BL / 8, 256, 0, stream>>>(x, gate_w, coef, lists, counts, probsum);
    moe_expert_ffn<<<dim3(BL / 32, E_), 256, 0, stream>>>(
        xb, w1t, w2t, b1, b2, coef, lists, counts, out);
    moe_aux<<<1, 32, 0, stream>>>(counts, probsum, out + (size_t)BL * D_);
}
